// DefConv_84774064488830
// MI455X (gfx1250) — compile-verified, matrix operand order checked by reading, not run on hardware
//
#include <hip/hip_runtime.h>

#define H_   56
#define W_   56
#define HW   3136
#define B_   8
#define NPIX 25088          // B_*HW
#define CIN  128
#define COUT 128
#define KTOT 1152           // CIN*9
#define KG   576            // per-group K = 64*9
#define NOFF 64             // 54 padded to 64
#define MT   64             // M rows per block
#define NBLK (NPIX / MT)    // 392
#define EPSV 1e-5f

typedef _Float16 v16h __attribute__((ext_vector_type(16)));
typedef _Float16 v8h  __attribute__((ext_vector_type(8)));
typedef float    v8f  __attribute__((ext_vector_type(8)));
typedef unsigned v4u  __attribute__((ext_vector_type(4)));
typedef int      v4i  __attribute__((ext_vector_type(4)));
typedef int      v8i  __attribute__((ext_vector_type(8)));

#if defined(__has_builtin)
#if __has_builtin(__builtin_amdgcn_tensor_load_to_lds) && __has_builtin(__builtin_amdgcn_s_wait_tensorcnt)
#define USE_TDM 1
#endif
#endif
#ifndef USE_TDM
#define USE_TDM 0
#endif

__device__ __forceinline__ v16h cat8(v8h a, v8h b) {
    return __builtin_shufflevector(a, b, 0,1,2,3,4,5,6,7,8,9,10,11,12,13,14,15);
}

#if USE_TDM
// TDM: DMA a [nrows x 32-half] tile of a [nrows x 1152-half] row-major f16 matrix
// into LDS (compacted rows). 4-byte elements: 16 dwords/row, row stride 576 dwords.
__device__ __forceinline__ void tdm_load_b_tile(const _Float16* gsrc, void* lds_dst, int nrows) {
    unsigned lds = (unsigned)(size_t)lds_dst;               // flat addr[31:0] == LDS offset
    unsigned long long ga = (unsigned long long)(size_t)gsrc;
    v4u g0;
    g0[0] = 1u;                                             // count=1, user descriptor
    g0[1] = lds;                                            // lds_addr
    g0[2] = (unsigned)(ga & 0xffffffffull);                 // global_addr[31:0]
    g0[3] = (unsigned)((ga >> 32) & 0x1ffffffull) | (2u << 30); // global_addr[56:32] | type=2
    v8i g1;
    g1[0] = (int)(2u << 16);                                // wg_mask=0, data_size=2 (4B)
    g1[1] = (int)(576u << 16);                              // tensor_dim0[15:0] @bits63:48
    g1[2] = (int)(((unsigned)nrows) << 16);                 // dim0 hi=0 | tensor_dim1 lo16
    g1[3] = (int)(16u << 16);                               // dim1 hi=0 | tile_dim0=16
    g1[4] = nrows;                                          // tile_dim1 | tile_dim2=0
    g1[5] = 576;                                            // tensor_dim0_stride lo32
    g1[6] = 0;                                              // stride hi | dim1_stride lo
    g1[7] = 0;
    v4i g2 = {0, 0, 0, 0};
    v4i g3 = {0, 0, 0, 0};
#if __clang_major__ >= 23
    v8i g4 = {0, 0, 0, 0, 0, 0, 0, 0};
    __builtin_amdgcn_tensor_load_to_lds(g0, g1, g2, g3, g4, 0);
#else
    __builtin_amdgcn_tensor_load_to_lds(g0, g1, g2, g3, 0);
#endif
}
#endif

// ---------------- K1: relu + f32->f16, NCHW -> NHWC ----------------
__global__ void k_relu_cvt(const float* __restrict__ x, _Float16* __restrict__ xrh) {
    int tid = blockIdx.x * blockDim.x + threadIdx.x;
    if (tid >= NPIX * CIN) return;
    int c   = tid & (CIN - 1);
    int pix = tid >> 7;
    int b   = pix / HW;
    int hw  = pix - b * HW;
    float v = x[(size_t)(b * CIN + c) * HW + hw];
    v = v > 0.f ? v : 0.f;
    xrh[(size_t)pix * CIN + c] = (_Float16)v;
}

// ---------------- K1b: pack offset weights B^T [64][1152], k = tap*128 + c ----------------
__global__ void k_pack_woff(const float* __restrict__ w_off, _Float16* __restrict__ bp) {
    int tid = blockIdx.x * blockDim.x + threadIdx.x;
    if (tid >= NOFF * KTOT) return;
    int n = tid / KTOT;
    int k = tid - n * KTOT;
    int c = k & 127;
    int t = k >> 7;
    float v = 0.f;
    if (n < 54) v = w_off[(size_t)(n * CIN + c) * 9 + t];
    bp[tid] = (_Float16)v;
}

// ---------------- K1c: pack main weights B^T [128][1152], k' = g*576 + p*64 + c ----------------
__global__ void k_pack_wmain(const float* __restrict__ w, _Float16* __restrict__ bp) {
    int tid = blockIdx.x * blockDim.x + threadIdx.x;
    if (tid >= COUT * KTOT) return;
    int n   = tid / KTOT;
    int k   = tid - n * KTOT;
    int g   = k / KG;
    int rem = k - g * KG;
    int p   = rem >> 6;
    int c   = rem & 63;
    bp[tid] = (_Float16)w[(size_t)n * KTOT + (g * 64 + c) * 9 + p];
}

// ---------------- K2: offset conv GEMM  M=NPIX, N=64, K=1152 ----------------
__global__ __launch_bounds__(128) void k_gemm_off(const _Float16* __restrict__ xrh,
                                                  const _Float16* __restrict__ bpack,
                                                  const float*    __restrict__ b_off,
                                                  float*          __restrict__ om) {
    __shared__ __align__(16) _Float16 As[MT * 32];
#if USE_TDM
    __shared__ __align__(16) _Float16 Bs[2][NOFF * 32];
#endif
    const int tid   = threadIdx.x;
    const int lane  = tid & 31;
    const int wv    = tid >> 5;
    const int pbase = blockIdx.x * MT;
    const int nbase = wv * 16;
    const int row   = lane & 15;
    const int kb    = (lane >> 4) << 3;

    const int arow = tid >> 1;
    const int klb  = (tid & 1) << 4;
    const int pix  = pbase + arow;
    const int bimg = pix / HW;
    const int hw   = pix - bimg * HW;
    const int hh   = hw / W_;
    const int ww   = hw - hh * W_;
    const _Float16* xb0 = xrh + (size_t)bimg * HW * CIN;

    v8f acc[4] = {{}, {}, {}, {}};

#if USE_TDM
    if (wv == 0) tdm_load_b_tile(bpack, &Bs[0][0], NOFF);
#endif

    for (int kc = 0; kc < KTOT; kc += 32) {
        int t  = kc >> 7;
        int ki = t / 3, kj = t - ki * 3;
        int cb = (kc & 127) + klb;
        int y  = hh - 1 + ki, x = ww - 1 + kj;
        v8h va = {}, vb = {};
        if (y >= 0 && y < H_ && x >= 0 && x < W_) {
            const _Float16* src = xb0 + (size_t)(y * W_ + x) * CIN + cb;
            va = *(const v8h*)src;
            vb = *(const v8h*)(src + 8);
        }
        *(v8h*)&As[arow * 32 + klb]     = va;
        *(v8h*)&As[arow * 32 + klb + 8] = vb;

#if USE_TDM
        const int buf = (kc >> 5) & 1;
        if (wv == 0) {
            if (kc + 32 < KTOT) {
                tdm_load_b_tile(bpack + kc + 32, &Bs[buf ^ 1][0], NOFF);
                __builtin_amdgcn_s_wait_tensorcnt(1);
            } else {
                __builtin_amdgcn_s_wait_tensorcnt(0);
            }
        }
#endif
        __syncthreads();

#if USE_TDM
        const _Float16* bsrc = &Bs[buf][(nbase + row) * 32 + kb];
#else
        const _Float16* bsrc = bpack + (size_t)(nbase + row) * KTOT + kc + kb;
        if (kc + 32 < KTOT) __builtin_prefetch(bsrc + 32, 0, 1);
#endif
        v16h bf = cat8(*(const v8h*)bsrc, *(const v8h*)(bsrc + 16));

#pragma unroll
        for (int s = 0; s < 4; ++s) {
            const _Float16* ap = &As[(s * 16 + row) * 32 + kb];
            v16h af = cat8(*(const v8h*)ap, *(const v8h*)(ap + 16));
            acc[s] = __builtin_amdgcn_wmma_f32_16x16x32_f16(false, af, false, bf,
                                                            (short)0, acc[s], false, false);
        }
        __syncthreads();
    }

    int n = nbase + row;
    float bias = (n < 54) ? b_off[n] : 0.f;
    int moff = (lane >> 4) << 3;
#pragma unroll
    for (int s = 0; s < 4; ++s) {
#pragma unroll
        for (int r = 0; r < 8; ++r) {
            int m = pbase + s * 16 + moff + r;
            om[(size_t)m * NOFF + n] = acc[s][r] + bias;
        }
    }
}

// ---------------- K2b: bilinear sampling params per (pixel, g, p) ----------------
__global__ void k_params(const float* __restrict__ om, uint4* __restrict__ prm) {
    int idx = blockIdx.x * blockDim.x + threadIdx.x;
    if (idx >= NPIX * 18) return;
    int pix = idx / 18;
    int gp  = idx - pix * 18;
    int p   = gp % 9;
    int ki  = p / 3, kj = p - ki * 3;
    int hw  = pix % HW;
    int h   = hw / W_, w = hw - h * W_;
    const float* o = om + (size_t)pix * NOFF;
    float oy = o[gp], ox = o[18 + gp], ml = o[36 + gp];
    float ys = oy + (float)(h - 1 + ki);
    float xs = ox + (float)(w - 1 + kj);
    float y0f = floorf(ys), x0f = floorf(xs);
    float wy = ys - y0f, wx = xs - x0f;
    int y0 = (int)y0f, x0 = (int)x0f;
    int y1 = y0 + 1,   x1 = x0 + 1;
    float msk = 1.f / (1.f + __expf(-ml));
    float vy0 = (y0 >= 0 && y0 < H_) ? 1.f : 0.f;
    float vy1 = (y1 >= 0 && y1 < H_) ? 1.f : 0.f;
    float vx0 = (x0 >= 0 && x0 < W_) ? 1.f : 0.f;
    float vx1 = (x1 >= 0 && x1 < W_) ? 1.f : 0.f;
    float w00 = (1.f - wy) * (1.f - wx) * msk * vy0 * vx0;
    float w01 = (1.f - wy) * wx         * msk * vy0 * vx1;
    float w10 = wy * (1.f - wx)         * msk * vy1 * vx0;
    float w11 = wy * wx                 * msk * vy1 * vx1;
    int y0c = min(max(y0, 0), H_ - 1), y1c = min(max(y1, 0), H_ - 1);
    int x0c = min(max(x0, 0), W_ - 1), x1c = min(max(x1, 0), W_ - 1);
    union { _Float16 hh[2]; unsigned u; } u0, u1;
    u0.hh[0] = (_Float16)w00; u0.hh[1] = (_Float16)w01;
    u1.hh[0] = (_Float16)w10; u1.hh[1] = (_Float16)w11;
    uint4 r;
    r.x = (unsigned)y0c | ((unsigned)x0c << 8) | ((unsigned)y1c << 16) | ((unsigned)x1c << 24);
    r.y = u0.u; r.z = u1.u; r.w = 0u;
    prm[idx] = r;
}

// ---------------- K3: main GEMM fused with bilinear gather  M=NPIX, N=128, K=1152 ----------------
__global__ __launch_bounds__(256) void k_gemm_main(const _Float16* __restrict__ xrh,
                                                   const uint4*    __restrict__ prm,
                                                   const _Float16* __restrict__ bpack,
                                                   const float*    __restrict__ bias,
                                                   float*          __restrict__ onh) {
    __shared__ __align__(16) _Float16 As[MT * 32];
#if USE_TDM
    __shared__ __align__(16) _Float16 Bs[2][COUT * 32];
#endif
    const int tid   = threadIdx.x;
    const int lane  = tid & 31;
    const int wv    = tid >> 5;
    const int pbase = blockIdx.x * MT;
    const int nbase = wv * 16;
    const int row   = lane & 15;
    const int kb    = (lane >> 4) << 3;

    const int arow = tid >> 2;
    const int klb  = (tid & 3) << 3;
    const int pix  = pbase + arow;
    const int bimg = pix / HW;
    const _Float16* xb0 = xrh + (size_t)bimg * HW * CIN;
    const uint4* prow = prm + (size_t)pix * 18;

    v8f acc[4] = {{}, {}, {}, {}};

#if USE_TDM
    if (wv == 0) tdm_load_b_tile(bpack, &Bs[0][0], COUT);
#endif

    for (int kc = 0; kc < KTOT; kc += 32) {
        int g   = kc / KG;
        int rem = kc - g * KG;
        int p   = rem >> 6;
        int cb  = g * 64 + (rem & 63) + klb;

        uint4 pr = prow[g * 9 + p];
        int y0c = pr.x & 255,         x0c = (pr.x >> 8) & 255;
        int y1c = (pr.x >> 16) & 255, x1c = (pr.x >> 24) & 255;
        const _Float16* pb = xb0 + cb;
        v8h h00 = *(const v8h*)(pb + (size_t)(y0c * W_ + x0c) * CIN);
        v8h h01 = *(const v8h*)(pb + (size_t)(y0c * W_ + x1c) * CIN);
        v8h h10 = *(const v8h*)(pb + (size_t)(y1c * W_ + x0c) * CIN);
        v8h h11 = *(const v8h*)(pb + (size_t)(y1c * W_ + x1c) * CIN);
        union { unsigned u; _Float16 hh[2]; } u0, u1;
        u0.u = pr.y; u1.u = pr.z;
        float w00 = (float)u0.hh[0], w01 = (float)u0.hh[1];
        float w10 = (float)u1.hh[0], w11 = (float)u1.hh[1];
        v8h res;
#pragma unroll
        for (int i = 0; i < 8; ++i) {
            float v = (float)h00[i] * w00 + (float)h01[i] * w01
                    + (float)h10[i] * w10 + (float)h11[i] * w11;
            res[i] = (_Float16)v;
        }
        *(v8h*)&As[arow * 32 + klb] = res;

#if USE_TDM
        const int buf = (kc >> 5) & 1;
        if (wv == 0) {
            if (kc + 32 < KTOT) {
                tdm_load_b_tile(bpack + kc + 32, &Bs[buf ^ 1][0], COUT);
                __builtin_amdgcn_s_wait_tensorcnt(1);
            } else {
                __builtin_amdgcn_s_wait_tensorcnt(0);
            }
        }
#endif
        __syncthreads();

#if USE_TDM
        const _Float16* bsrc = &Bs[buf][(nbase + row) * 32 + kb];
#else
        const _Float16* bsrc = bpack + (size_t)(nbase + row) * KTOT + kc + kb;
        if (kc + 32 < KTOT) __builtin_prefetch(bsrc + 32, 0, 1);
#endif
        v16h bf = cat8(*(const v8h*)bsrc, *(const v8h*)(bsrc + 16));

#pragma unroll
        for (int s = 0; s < 4; ++s) {
            const _Float16* ap = &As[(s * 16 + row) * 32 + kb];
            v16h af = cat8(*(const v8h*)ap, *(const v8h*)(ap + 16));
            acc[s] = __builtin_amdgcn_wmma_f32_16x16x32_f16(false, af, false, bf,
                                                            (short)0, acc[s], false, false);
        }
        __syncthreads();
    }

    int n = nbase + row;
    float bs = bias[n];
    int moff = (lane >> 4) << 3;
#pragma unroll
    for (int s = 0; s < 4; ++s) {
#pragma unroll
        for (int r = 0; r < 8; ++r) {
            int m = pbase + s * 16 + moff + r;
            onh[(size_t)m * COUT + n] = acc[s][r] + bs;
        }
    }
}

// ---------------- K4a: coalesced partial sums per (block, channel) ----------------
__global__ __launch_bounds__(256) void k_stats1(const float* __restrict__ onh,
                                                float* __restrict__ ps, float* __restrict__ ps2) {
    __shared__ float s1[256];
    __shared__ float s2[256];
    int tid = threadIdx.x;
    int c   = tid & 127;
    int r   = tid >> 7;
    int pb  = blockIdx.x * MT;
    float a = 0.f, b = 0.f;
    for (int i = r; i < MT; i += 2) {
        float v = onh[(size_t)(pb + i) * COUT + c];
        a += v; b += v * v;
    }
    s1[tid] = a; s2[tid] = b;
    __syncthreads();
    if (r == 0) {
        ps [blockIdx.x * 128 + c] = s1[tid] + s1[tid + 128];
        ps2[blockIdx.x * 128 + c] = s2[tid] + s2[tid + 128];
    }
}

// ---------------- K4b: finalize mean / rsqrt(var+eps), deterministic ----------------
__global__ __launch_bounds__(128) void k_stats2(const float* __restrict__ ps,
                                                const float* __restrict__ ps2,
                                                float* __restrict__ stats) {
    int ch = threadIdx.x;
    float a = 0.f, b = 0.f;
    for (int i = 0; i < NBLK; ++i) {
        a += ps [i * 128 + ch];
        b += ps2[i * 128 + ch];
    }
    float mu  = a / (float)NPIX;
    float var = b / (float)NPIX - mu * mu;
    stats[ch]        = mu;
    stats[COUT + ch] = rsqrtf(var + EPSV);
}

// ---------------- K5: normalize + gamma/beta, NHWC -> NCHW out ----------------
__global__ void k_norm(const float* __restrict__ onh, const float* __restrict__ stats,
                       const float* __restrict__ gamma, const float* __restrict__ beta,
                       float* __restrict__ out) {
    int tid = blockIdx.x * blockDim.x + threadIdx.x;
    if (tid >= NPIX * COUT) return;
    int ch  = tid & (COUT - 1);
    int pix = tid >> 7;
    float v = onh[tid];
    float y = (v - stats[ch]) * stats[COUT + ch] * gamma[ch] + beta[ch];
    int b  = pix / HW;
    int hw = pix - b * HW;
    out[(size_t)(b * COUT + ch) * HW + hw] = y;
}

// ---------------- workspace layout (bytes) ----------------
#define WS_XRH    ((size_t)0)
#define WS_BPOFF  (WS_XRH   + (size_t)NPIX * CIN * 2)
#define WS_BPMAIN (WS_BPOFF + (size_t)NOFF * KTOT * 2)
#define WS_OM     (WS_BPMAIN+ (size_t)COUT * KTOT * 2)
#define WS_PRM    (WS_OM    + (size_t)NPIX * NOFF * 4)
#define WS_ONH    (WS_PRM   + (size_t)NPIX * 18 * 16)
#define WS_STATS  (WS_ONH   + (size_t)NPIX * COUT * 4)
#define WS_PS     (WS_STATS + (size_t)2 * COUT * 4)
#define WS_PS2    (WS_PS    + (size_t)NBLK * 128 * 4)

extern "C" void kernel_launch(void* const* d_in, const int* in_sizes, int n_in,
                              void* d_out, int out_size, void* d_ws, size_t ws_size,
                              hipStream_t stream) {
    (void)in_sizes; (void)n_in; (void)out_size; (void)ws_size;
    const float* x     = (const float*)d_in[0];
    const float* w_off = (const float*)d_in[1];
    const float* b_off = (const float*)d_in[2];
    const float* w     = (const float*)d_in[3];
    const float* b     = (const float*)d_in[4];
    const float* gamma = (const float*)d_in[5];
    const float* beta  = (const float*)d_in[6];
    float* out = (float*)d_out;
    char*  ws  = (char*)d_ws;

    _Float16* xrh    = (_Float16*)(ws + WS_XRH);
    _Float16* bpoff  = (_Float16*)(ws + WS_BPOFF);
    _Float16* bpmain = (_Float16*)(ws + WS_BPMAIN);
    float*    om     = (float*)   (ws + WS_OM);
    uint4*    prm    = (uint4*)   (ws + WS_PRM);
    float*    onh    = (float*)   (ws + WS_ONH);
    float*    stats  = (float*)   (ws + WS_STATS);
    float*    ps     = (float*)   (ws + WS_PS);
    float*    ps2    = (float*)   (ws + WS_PS2);

    k_relu_cvt  <<<(NPIX * CIN + 255) / 256, 256, 0, stream>>>(x, xrh);
    k_pack_woff <<<(NOFF * KTOT + 255) / 256, 256, 0, stream>>>(w_off, bpoff);
    k_pack_wmain<<<(COUT * KTOT + 255) / 256, 256, 0, stream>>>(w, bpmain);
    k_gemm_off  <<<NBLK, 128, 0, stream>>>(xrh, bpoff, b_off, om);
    k_params    <<<(NPIX * 18 + 255) / 256, 256, 0, stream>>>(om, prm);
    k_gemm_main <<<NBLK, 256, 0, stream>>>(xrh, prm, bpmain, b, onh);
    k_stats1    <<<NBLK, 256, 0, stream>>>(onh, ps, ps2);
    k_stats2    <<<1, 128, 0, stream>>>(ps, ps2, stats);
    k_norm      <<<(NPIX * COUT + 255) / 256, 256, 0, stream>>>(onh, stats, gamma, beta, out);
}
